// AGNNLayer_80126909874572
// MI455X (gfx1250) — matrix operand order repaired and checked by reading, not compiled, not run on hardware
//
#include <hip/hip_runtime.h>
#include <hip/hip_bf16.h>

typedef __attribute__((ext_vector_type(16))) __bf16 v16bf;
typedef __attribute__((ext_vector_type(8)))  __bf16 v8bf;
typedef __attribute__((ext_vector_type(8)))  float  v8f;
typedef __attribute__((ext_vector_type(4)))  float  v4f;

// ---------------------------------------------------------------------------
// WMMA helper: D = A(16x32 bf16) * B(32x16 bf16) + C(16x16 f32)
// ---------------------------------------------------------------------------
__device__ __forceinline__ v8f wmma_bf16(v16bf a, v16bf b, v8f c) {
  return __builtin_amdgcn_wmma_f32_16x16x32_bf16(
      /*neg_a=*/false, a, /*neg_b=*/false, b,
      /*c_mod=*/(short)0, c, /*reuse_a=*/false, /*reuse_b=*/false);
}

// A fragment (16x32, bf16) from row-major f32, lda = 128.
// ISA layout: lane L holds row M=L&15; lanes 0-15: K = k0+{0..7, 16..23},
// lanes 16-31: K = k0+{8..15, 24..31}. Element pairs map to VGPRs in order.
__device__ __forceinline__ v16bf load_A_frag_f32(const float* __restrict__ A,
                                                 long row, int k0, int half) {
  const float* p = A + row * 128 + k0 + half * 8;
  v4f x0 = *(const v4f*)(p);
  v4f x1 = *(const v4f*)(p + 4);
  v4f x2 = *(const v4f*)(p + 16);
  v4f x3 = *(const v4f*)(p + 20);
  v16bf a;
#pragma unroll
  for (int i = 0; i < 4; ++i) {
    a[i]      = (__bf16)x0[i];
    a[4 + i]  = (__bf16)x1[i];
    a[8 + i]  = (__bf16)x2[i];
    a[12 + i] = (__bf16)x3[i];
  }
  return a;
}

// B fragment (32x16 bf16), pre-packed in fragment order by prep_weights:
// contiguous 16 bf16 per (ktile, ctile, lane). Two b128 loads.
__device__ __forceinline__ v16bf load_B_frag(const __bf16* __restrict__ Wf,
                                             int kt, int ct, int lane) {
  const __bf16* p = Wf + (((size_t)((kt * 8 + ct) * 32 + lane)) << 4);
  v8bf lo = *(const v8bf*)p;
  v8bf hi = *(const v8bf*)(p + 8);
  v16bf b;
#pragma unroll
  for (int i = 0; i < 8; ++i) { b[i] = lo[i]; b[8 + i] = hi[i]; }
  return b;
}

// 4 WMMAs over K=128 with all B fragments preloaded before the MMA chain,
// so the scheduler can overlap loads with math instead of wait-0 stalls.
__device__ __forceinline__ v8f gemm_k128(const v16bf a[4],
                                         const __bf16* __restrict__ Wf,
                                         int ct, int lane) {
  v16bf bfr[4];
#pragma unroll
  for (int kt = 0; kt < 4; ++kt) bfr[kt] = load_B_frag(Wf, kt, ct, lane);
  v8f acc = {0, 0, 0, 0, 0, 0, 0, 0};
#pragma unroll
  for (int kt = 0; kt < 4; ++kt) acc = wmma_bf16(a[kt], bfr[kt], acc);
  return acc;
}

// A fragment of z = BN(e_hat) built on the fly: z = ehat*sc[c] + sh[c]
__device__ __forceinline__ v16bf load_z_frag(const float* __restrict__ eh,
                                             long row, int k0, int half,
                                             const float* sc, const float* sh) {
  const float* p = eh + row * 128 + k0 + half * 8;
  v4f x0 = *(const v4f*)(p);
  v4f x1 = *(const v4f*)(p + 4);
  v4f x2 = *(const v4f*)(p + 16);
  v4f x3 = *(const v4f*)(p + 20);
  int c0 = k0 + half * 8;
  v16bf a;
#pragma unroll
  for (int i = 0; i < 4; ++i) {
    a[i]      = (__bf16)(x0[i] * sc[c0 + i]      + sh[c0 + i]);
    a[4 + i]  = (__bf16)(x1[i] * sc[c0 + 4 + i]  + sh[c0 + 4 + i]);
    a[8 + i]  = (__bf16)(x2[i] * sc[c0 + 16 + i] + sh[c0 + 16 + i]);
    a[12 + i] = (__bf16)(x3[i] * sc[c0 + 20 + i] + sh[c0 + 20 + i]);
  }
  return a;
}

// A fragment from LDS-resident bf16 row-major tile (16x128), two b128 ds loads.
__device__ __forceinline__ v16bf load_t_frag(const __bf16* tb, int r, int k0, int half) {
  const __bf16* p = tb + r * 128 + k0 + half * 8;
  v8bf lo = *(const v8bf*)p;
  v8bf hi = *(const v8bf*)(p + 16);
  v16bf a;
#pragma unroll
  for (int i = 0; i < 8; ++i) { a[i] = lo[i]; a[8 + i] = hi[i]; }
  return a;
}

// ---------------------------------------------------------------------------
// Weight prep: 7 D x D f32 weights -> bf16 in B-fragment lane order.
// b[i] = W[(kt*32 + half*8 + (i&7) + (i>>3)*16)*128 + ct*16 + (lane&15)]
// ---------------------------------------------------------------------------
__global__ void __launch_bounds__(256) prep_weights(
    const float* __restrict__ w0, const float* __restrict__ w1,
    const float* __restrict__ w2, const float* __restrict__ w3,
    const float* __restrict__ w4, const float* __restrict__ w5,
    const float* __restrict__ w6, __bf16* __restrict__ out) {
  const float* ws[7] = {w0, w1, w2, w3, w4, w5, w6};
  const float* W = ws[blockIdx.x];
  __bf16* o = out + (size_t)blockIdx.x * 16384;
  for (int idx = threadIdx.x; idx < 1024; idx += 256) {  // 4 kt * 8 ct * 32 lanes
    int lane = idx & 31;
    int ct = (idx >> 5) & 7;
    int kt = idx >> 8;
    int nn = lane & 15, half = lane >> 4;
    __bf16* d = o + ((size_t)idx << 4);
#pragma unroll
    for (int i = 0; i < 16; ++i)
      d[i] = (__bf16)W[(kt * 32 + half * 8 + (i & 7) + (i >> 3) * 16) * 128 + ct * 16 + nn];
  }
}

// ---------------------------------------------------------------------------
// Node GEMMs: out = h @ W for W in {Qw, Rw, Uw, Vw} (blockIdx.y selects)
// ---------------------------------------------------------------------------
__global__ void __launch_bounds__(256) node_gemm(
    const float* __restrict__ h, const __bf16* __restrict__ wfrag,
    float* __restrict__ o0, float* __restrict__ o1,
    float* __restrict__ o2, float* __restrict__ o3, int N) {
  int y = blockIdx.y;
  const __bf16* Wf = wfrag + (size_t)(y + 1) * 16384;  // frags: P,Q,R,U,V,W1,W2
  float* out = (y == 0) ? o0 : (y == 1) ? o1 : (y == 2) ? o2 : o3;
  int tid = threadIdx.x, wave = tid >> 5, lane = tid & 31;
  int nn = lane & 15, mh = lane >> 4, half = mh;
  long row0 = (long)blockIdx.x * 128 + wave * 16;
  if (row0 >= N) return;
  bool full = (row0 + 16 <= (long)N);  // wave-uniform
  long arow = row0 + nn; if (arow >= N) arow = N - 1;
  v16bf a[4];
#pragma unroll
  for (int kt = 0; kt < 4; ++kt) a[kt] = load_A_frag_f32(h, arow, kt * 32, half);
#pragma unroll
  for (int ct = 0; ct < 8; ++ct) {
    v8f acc = gemm_k128(a, Wf, ct, lane);
    int col = ct * 16 + nn;
    if (full) {
      float* p = out + (row0 + mh * 8) * 128 + col;
#pragma unroll
      for (int i = 0; i < 8; ++i) p[i * 128] = acc[i];
    } else {
#pragma unroll
      for (int i = 0; i < 8; ++i) {
        long row = row0 + mh * 8 + i;
        if (row < N) out[row * 128 + col] = acc[i];
      }
    }
  }
}

// ---------------------------------------------------------------------------
// Edge pass 1: e_hat = e@Pw + Hq[src] + Hr[dst]; BN stats; gate scatter
// ---------------------------------------------------------------------------
__global__ void __launch_bounds__(256) edge_pass1(
    const float* __restrict__ e, const float* __restrict__ Hq,
    const float* __restrict__ Hr, const float* __restrict__ Hv,
    const int* __restrict__ src, const int* __restrict__ dst,
    const __bf16* __restrict__ Pf, float* __restrict__ ehat,
    float* __restrict__ agg, float* __restrict__ gsum,
    float* __restrict__ gssq, int E) {
  __shared__ float s_sum[128];
  __shared__ float s_ssq[128];
  int tid = threadIdx.x;
  if (tid < 128) { s_sum[tid] = 0.f; s_ssq[tid] = 0.f; }
  __syncthreads();
  int wave = tid >> 5, lane = tid & 31;
  int nn = lane & 15, mh = lane >> 4, half = mh;
  long row0 = (long)blockIdx.x * 128 + wave * 16;
  bool full = (row0 + 16 <= (long)E);  // wave-uniform
  long arow = row0 + nn; if (arow >= E) arow = E - 1;
  v16bf a[4];
#pragma unroll
  for (int kt = 0; kt < 4; ++kt) a[kt] = load_A_frag_f32(e, arow, kt * 32, half);
  int s8[8], d8[8];
#pragma unroll
  for (int i = 0; i < 8; ++i) {
    long rr = row0 + mh * 8 + i; if (rr >= E) rr = E - 1;
    s8[i] = src[rr]; d8[i] = dst[rr];
  }
#pragma unroll
  for (int ct = 0; ct < 8; ++ct) {
    v8f acc = gemm_k128(a, Pf, ct, lane);
    int col = ct * 16 + nn;
    float ls = 0.f, lq = 0.f;
    if (full) {
      float* pe = ehat + (row0 + mh * 8) * 128 + col;
#pragma unroll
      for (int i = 0; i < 8; ++i) {
        float v = acc[i] + Hq[(long)s8[i] * 128 + col] + Hr[(long)d8[i] * 128 + col];
        pe[i * 128] = v;
        ls += v; lq += v * v;
        float g = 1.f / (1.f + __expf(-v));
        atomicAdd(&agg[(long)s8[i] * 128 + col], g * Hv[(long)d8[i] * 128 + col]);
      }
    } else {
#pragma unroll
      for (int i = 0; i < 8; ++i) {
        long row = row0 + mh * 8 + i;
        if (row < E) {
          float v = acc[i] + Hq[(long)s8[i] * 128 + col] + Hr[(long)d8[i] * 128 + col];
          ehat[row * 128 + col] = v;
          ls += v; lq += v * v;
          float g = 1.f / (1.f + __expf(-v));
          atomicAdd(&agg[(long)s8[i] * 128 + col], g * Hv[(long)d8[i] * 128 + col]);
        }
      }
    }
    atomicAdd(&s_sum[col], ls);
    atomicAdd(&s_ssq[col], lq);
  }
  __syncthreads();
  if (tid < 128) { atomicAdd(&gsum[tid], s_sum[tid]); atomicAdd(&gssq[tid], s_ssq[tid]); }
}

// ---------------------------------------------------------------------------
// Finalize BN stats: s = gamma*rsqrt(var+eps), t = beta - mu*s
// ---------------------------------------------------------------------------
__global__ void finalize_stats(const float* __restrict__ sum,
                               const float* __restrict__ ssq,
                               const float* __restrict__ gamma,
                               const float* __restrict__ beta, float cnt,
                               float* __restrict__ s, float* __restrict__ t) {
  int c = threadIdx.x;
  float mu = sum[c] / cnt;
  float var = ssq[c] / cnt - mu * mu;
  float sc = gamma[c] * rsqrtf(var + 1e-5f);
  s[c] = sc;
  t[c] = beta[c] - mu * sc;
}

// ---------------------------------------------------------------------------
// Edge pass 2: e_new = e + relu(BN(e_hat)@W1 + b1)@W2 + b2  (fused via LDS)
// ---------------------------------------------------------------------------
__global__ void __launch_bounds__(256) edge_pass2(
    const float* __restrict__ e, const float* __restrict__ ehat,
    const __bf16* __restrict__ W1f, const __bf16* __restrict__ W2f,
    const float* __restrict__ b1, const float* __restrict__ b2,
    const float* __restrict__ se, const float* __restrict__ te,
    float* __restrict__ eout, int E) {
  __shared__ float sc[128], sh[128], sb1[128], sb2[128];
  __shared__ __bf16 tbuf[8][16 * 128] __attribute__((aligned(16)));
  int tid = threadIdx.x;
  if (tid < 128) { sc[tid] = se[tid]; sh[tid] = te[tid]; sb1[tid] = b1[tid]; sb2[tid] = b2[tid]; }
  __syncthreads();
  int wave = tid >> 5, lane = tid & 31;
  int nn = lane & 15, mh = lane >> 4, half = mh;
  long row0 = (long)blockIdx.x * 128 + wave * 16;
  bool full = (row0 + 16 <= (long)E);  // wave-uniform
  long arow = row0 + nn; if (arow >= E) arow = E - 1;
  v16bf az[4];
#pragma unroll
  for (int kt = 0; kt < 4; ++kt) az[kt] = load_z_frag(ehat, arow, kt * 32, half, sc, sh);
  __bf16* tb = &tbuf[wave][0];
#pragma unroll
  for (int ct = 0; ct < 8; ++ct) {
    v8f acc = gemm_k128(az, W1f, ct, lane);
    int col = ct * 16 + nn;
    __bf16* pt = tb + mh * 8 * 128 + col;
#pragma unroll
    for (int i = 0; i < 8; ++i)
      pt[i * 128] = (__bf16)fmaxf(acc[i] + sb1[col], 0.f);
  }
  __syncthreads();
  v16bf at[4];
#pragma unroll
  for (int kt = 0; kt < 4; ++kt) at[kt] = load_t_frag(tb, nn, kt * 32, half);
#pragma unroll
  for (int ct = 0; ct < 8; ++ct) {
    v8f acc = gemm_k128(at, W2f, ct, lane);
    int col = ct * 16 + nn;
    if (full) {
      const float* pe = e + (row0 + mh * 8) * 128 + col;
      float* po = eout + (row0 + mh * 8) * 128 + col;
#pragma unroll
      for (int i = 0; i < 8; ++i)
        po[i * 128] = pe[i * 128] + acc[i] + sb2[col];
    } else {
#pragma unroll
      for (int i = 0; i < 8; ++i) {
        long row = row0 + mh * 8 + i;
        if (row < E) eout[row * 128 + col] = e[row * 128 + col] + acc[i] + sb2[col];
      }
    }
  }
}

// ---------------------------------------------------------------------------
// Node BN stats over x = Hu + agg (float4 vectorized)
// ---------------------------------------------------------------------------
__global__ void __launch_bounds__(256) node_stats(
    const float* __restrict__ Hu, const float* __restrict__ agg,
    float* __restrict__ nsum, float* __restrict__ nssq, int N) {
  __shared__ float ss[128], sq[128];
  int tid = threadIdx.x;
  if (tid < 128) { ss[tid] = 0.f; sq[tid] = 0.f; }
  __syncthreads();
  int col4 = (tid & 31) * 4;   // column group of 4
  int rh = tid >> 5;           // 0..7
  long row0 = (long)blockIdx.x * 64;
  v4f ls = {0, 0, 0, 0}, lq = {0, 0, 0, 0};
  for (int r = rh; r < 64; r += 8) {
    long row = row0 + r;
    if (row < N) {
      long idx = row * 128 + col4;
      v4f x = *(const v4f*)(Hu + idx);
      v4f y = *(const v4f*)(agg + idx);
      v4f v = x + y;
      ls += v; lq += v * v;
    }
  }
#pragma unroll
  for (int j = 0; j < 4; ++j) {
    atomicAdd(&ss[col4 + j], ls[j]);
    atomicAdd(&sq[col4 + j], lq[j]);
  }
  __syncthreads();
  if (tid < 128) { atomicAdd(&nsum[tid], ss[tid]); atomicAdd(&nssq[tid], sq[tid]); }
}

// ---------------------------------------------------------------------------
// h_new = h + alpha * ((Hu+agg)*s_n[c] + t_n[c])  (float4 vectorized)
// ---------------------------------------------------------------------------
__global__ void __launch_bounds__(256) node_apply(
    const float* __restrict__ h, const float* __restrict__ Hu,
    const float* __restrict__ agg, const float* __restrict__ sn,
    const float* __restrict__ tn, const float* __restrict__ alpha,
    float* __restrict__ hout, long total4) {
  float a = *alpha;
  for (long q = (long)blockIdx.x * blockDim.x + threadIdx.x; q < total4;
       q += (long)gridDim.x * blockDim.x) {
    long i = q * 4;
    int c = (int)(i & 127);
    v4f x = *(const v4f*)(Hu + i);
    v4f y = *(const v4f*)(agg + i);
    v4f hh = *(const v4f*)(h + i);
    v4f s = *(const v4f*)(sn + c);
    v4f t = *(const v4f*)(tn + c);
    v4f v = x + y;
    v4f r = hh + a * (v * s + t);
    *(v4f*)(hout + i) = r;
  }
}

// ---------------------------------------------------------------------------
extern "C" void kernel_launch(void* const* d_in, const int* in_sizes, int n_in,
                              void* d_out, int out_size, void* d_ws, size_t ws_size,
                              hipStream_t stream) {
  (void)n_in; (void)out_size; (void)ws_size;
  const float* h   = (const float*)d_in[0];
  const float* e   = (const float*)d_in[1];
  const int* eidx  = (const int*)d_in[2];
  const float* Pw  = (const float*)d_in[3];
  const float* Qw  = (const float*)d_in[4];
  const float* Rw  = (const float*)d_in[5];
  const float* Uw  = (const float*)d_in[6];
  const float* Vw  = (const float*)d_in[7];
  const float* W1  = (const float*)d_in[8];
  const float* b1  = (const float*)d_in[9];
  const float* W2  = (const float*)d_in[10];
  const float* b2  = (const float*)d_in[11];
  const float* ge  = (const float*)d_in[12];
  const float* be  = (const float*)d_in[13];
  const float* gn  = (const float*)d_in[14];
  const float* bn  = (const float*)d_in[15];
  const float* alpha = (const float*)d_in[16];

  long N = in_sizes[0] / 128;
  long E = in_sizes[2] / 2;
  const int* src = eidx;
  const int* dst = eidx + E;

  char* ws = (char*)d_ws;
  size_t off = 0;
  auto take = [&](size_t bytes) -> char* {
    char* p = ws + off;
    off = (off + bytes + 255) & ~(size_t)255;
    return p;
  };
  float*  ehat  = (float*)take((size_t)E * 128 * 4);
  float*  Hq    = (float*)take((size_t)N * 128 * 4);
  float*  Hr    = (float*)take((size_t)N * 128 * 4);
  float*  Hu    = (float*)take((size_t)N * 128 * 4);
  float*  Hv    = (float*)take((size_t)N * 128 * 4);
  float*  agg   = (float*)take((size_t)N * 128 * 4);
  __bf16* wfrag = (__bf16*)take((size_t)7 * 16384 * 2);
  float*  stats = (float*)take((size_t)8 * 128 * 4);
  float *esum = stats,       *essq = stats + 128;
  float *nsum = stats + 256, *nssq = stats + 384;
  float *se   = stats + 512, *te   = stats + 640;
  float *sn   = stats + 768, *tn   = stats + 896;

  // Buffers accumulated atomically must be zeroed every launch.
  hipMemsetAsync(agg, 0, (size_t)N * 128 * 4, stream);
  hipMemsetAsync(stats, 0, (size_t)4 * 128 * 4, stream);

  prep_weights<<<dim3(7), dim3(256), 0, stream>>>(Pw, Qw, Rw, Uw, Vw, W1, W2, wfrag);

  dim3 ng((unsigned)((N + 127) / 128), 4);
  node_gemm<<<ng, 256, 0, stream>>>(h, wfrag, Hq, Hr, Hu, Hv, (int)N);

  edge_pass1<<<dim3((unsigned)((E + 127) / 128)), 256, 0, stream>>>(
      e, Hq, Hr, Hv, src, dst, wfrag, ehat, agg, esum, essq, (int)E);

  finalize_stats<<<1, 128, 0, stream>>>(esum, essq, ge, be, (float)E, se, te);

  float* hout = (float*)d_out;
  float* eout = hout + (size_t)N * 128;
  edge_pass2<<<dim3((unsigned)((E + 127) / 128)), 256, 0, stream>>>(
      e, ehat, wfrag + (size_t)5 * 16384, wfrag + (size_t)6 * 16384,
      b1, b2, se, te, eout, (int)E);

  node_stats<<<dim3((unsigned)((N + 63) / 64)), 256, 0, stream>>>(
      Hu, agg, nsum, nssq, (int)N);
  finalize_stats<<<1, 128, 0, stream>>>(nsum, nssq, gn, bn, (float)N, sn, tn);

  long total4 = (N * 128) / 4;
  node_apply<<<dim3((unsigned)((total4 + 255) / 256)), 256, 0, stream>>>(
      h, Hu, agg, sn, tn, alpha, hout, total4);
}
